// ExternalEmbeddingAttention_55121610277510
// MI455X (gfx1250) — matrix operand order repaired and checked by reading, not compiled, not run on hardware
//
#include <hip/hip_runtime.h>
#include <cstdint>
#include <cstddef>

// CDNA5 / gfx1250: wave32, WMMA 16x16x32 bf16 (f32 accumulate),
// double-buffered LDS with TDM (tensor_load_to_lds) fetching both tiles.
typedef __attribute__((ext_vector_type(16))) __bf16 v16bf;
typedef __attribute__((ext_vector_type(8)))  float  v8f;
typedef unsigned int uint32;
typedef __attribute__((ext_vector_type(4))) unsigned int u32x4;
typedef __attribute__((ext_vector_type(8))) int i32x8;
typedef __attribute__((ext_vector_type(4))) int i32x4;

__device__ __forceinline__ uint32 bfbits(float f) {
  uint32 u = __builtin_bit_cast(uint32, f);
  return (u + 0x7FFFu + ((u >> 16) & 1u)) >> 16;   // RNE truncate to bf16
}
__device__ __forceinline__ __bf16 f2bf(float f) {
  unsigned short s = (unsigned short)bfbits(f);
  __bf16 b;
  __builtin_memcpy(&b, &s, 2);
  return b;
}

#if __has_builtin(__builtin_amdgcn_tensor_load_to_lds)
#define GEMM_USE_TDM 1
#else
#define GEMM_USE_TDM 0
#endif

// ---------------------------------------------------------------------------
// f32 -> bf16 bulk conversion (4 elems/thread, packed u32 stores)
// ---------------------------------------------------------------------------
__global__ __launch_bounds__(256) void to_bf16_kernel(
    const float* __restrict__ x, __bf16* __restrict__ y, int n4) {
  const int i = blockIdx.x * 256 + threadIdx.x;
  if (i >= n4) return;
  const float4 v = reinterpret_cast<const float4*>(x)[i];
  uint32* yo = reinterpret_cast<uint32*>(y);
  yo[2 * i + 0] = bfbits(v.x) | (bfbits(v.y) << 16);
  yo[2 * i + 1] = bfbits(v.z) | (bfbits(v.w) << 16);
}

// ---------------------------------------------------------------------------
// One-time weight transform: W[K][N] f32 -> Wt[N][K] bf16 (32x32 LDS tiles).
// Makes the GEMM B tile a plain 2D strided tile -> TDM-fetchable, and kills
// all per-K-step transpose VALU in the GEMM inner loop.
// ---------------------------------------------------------------------------
__global__ __launch_bounds__(256) void transpose_to_bf16(
    const float* __restrict__ W, __bf16* __restrict__ Wt, int K, int N) {
  __shared__ float t[32][33];
  const int k0 = blockIdx.y * 32;
  const int n0 = blockIdx.x * 32;
  const int tx = threadIdx.x & 31;
  const int ty = threadIdx.x >> 5;   // 0..7
#pragma unroll
  for (int r = 0; r < 32; r += 8)
    t[ty + r][tx] = W[(size_t)(k0 + ty + r) * N + n0 + tx];
  __syncthreads();
#pragma unroll
  for (int r = 0; r < 32; r += 8)
    Wt[(size_t)(n0 + ty + r) * K + k0 + tx] = f2bf(t[tx][ty + r]);
}

// ---------------------------------------------------------------------------
// TDM: DMA one (rows x 32) bf16 tile (row stride = K elems) into LDS.
// D# per CDNA5 ISA ch.8: group0 {count=1, lds_addr, global_addr, type=2},
// group1 {data_size=2B, pad 1 DWORD per 16 DWORDs -> [rows][34] layout,
//         tensor dims, tile 32 x rows, dim0 stride = K}.
// ---------------------------------------------------------------------------
__device__ __forceinline__ void tdm_load_tile(const __bf16* gsrc,
                                              unsigned lds_off, int K, int rows) {
#if GEMM_USE_TDM
  const unsigned long long ga = (unsigned long long)(uintptr_t)gsrc;
  u32x4 g0;
  g0[0] = 1u;                                               // count=1 (valid)
  g0[1] = lds_off;                                          // LDS byte address
  g0[2] = (uint32)(ga & 0xFFFFFFFFu);                       // global addr lo
  g0[3] = ((uint32)(ga >> 32) & 0x01FFFFFFu) | (2u << 30);  // addr hi | type=2
  i32x8 g1;
  g1[0] = (int)((1u << 16) | (1u << 20) | (3u << 22));  // 2B, pad_en, 16dw/1dw
  g1[1] = (int)(32u << 16);                  // tensor_dim0 = 32
  g1[2] = (int)((unsigned)rows << 16);       // tensor_dim1 = rows
  g1[3] = (int)(32u << 16);                  // tile_dim0 = 32
  g1[4] = rows;                              // tile_dim1 = rows
  g1[5] = K;                                 // tensor_dim0_stride
  g1[6] = 0;
  g1[7] = 0;
  const i32x4 z4 = {0, 0, 0, 0};
#if __has_include(<hip/amd_detail/amd_gfx1250_TDM.h>)
  const i32x8 z8 = {0, 0, 0, 0, 0, 0, 0, 0};
  __builtin_amdgcn_tensor_load_to_lds(g0, g1, z4, z4, z8, 0);
#else
  __builtin_amdgcn_tensor_load_to_lds(g0, g1, z4, z4, 0);
#endif
#endif
}

// ---------------------------------------------------------------------------
// GEMM: C[M,N] = Ab[M,K](bf16) @ Wt[N,K](bf16, pre-transposed) + bias (+epi)
// EPI: 0 = bias, 1 = bias+exact GELU, 2 = bias+residual(f32)
// OBF: 0 = f32 output, 1 = bf16 output (feeds a downstream GEMM)
// Block tile 64x128x32, 8 waves, wave tile 16x64. Double-buffered LDS;
// wave0 issues TDM for tile i+1 before the WMMAs of tile i.
// ---------------------------------------------------------------------------
template <int EPI, int OBF>
__global__ __launch_bounds__(256) void gemm_bf16_wmma(
    const __bf16* __restrict__ Ab, const __bf16* __restrict__ Wt,
    const float* __restrict__ bias, const float* __restrict__ res,
    void* __restrict__ Cout, int M, int N, int K) {
  __shared__ __bf16 As[2][64][34];    // [buf][m][k], 1-DWORD row pad (TDM pad)
  __shared__ __bf16 Bs[2][128][34];   // [buf][n][k]

  const int tid   = threadIdx.x;
  const int lane  = tid & 31;
  const int wave  = tid >> 5;
  const int waveM = wave & 3;
  const int waveN = wave >> 2;
  const int bm0   = blockIdx.y * 64;
  const int bn0   = blockIdx.x * 128;

  constexpr bool kUseTdm = (GEMM_USE_TDM != 0);
  unsigned ldsA[2], ldsB[2];
  ldsA[0] = (unsigned)(uintptr_t)(&As[0][0][0]);
  ldsA[1] = (unsigned)(uintptr_t)(&As[1][0][0]);
  ldsB[0] = (unsigned)(uintptr_t)(&Bs[0][0][0]);
  ldsB[1] = (unsigned)(uintptr_t)(&Bs[1][0][0]);

  const v8f vzero = {0.f, 0.f, 0.f, 0.f, 0.f, 0.f, 0.f, 0.f};
  v8f acc[4];
#pragma unroll
  for (int i = 0; i < 4; ++i) acc[i] = vzero;

  // manual fallback staging maps (contiguous row copies)
  const int ar = tid >> 2;            // 0..63
  const int ac = (tid & 3) * 8;       // 0,8,16,24
  const int br = tid >> 1;            // 0..127
  const int bc = (tid & 1) * 16;      // 0,16

  auto stage = [&](int buf, int kb) {
    if (kUseTdm) {
      if (wave == 0) {
        tdm_load_tile(Ab + (size_t)bm0 * K + kb, ldsA[buf], K, 64);
        tdm_load_tile(Wt + (size_t)bn0 * K + kb, ldsB[buf], K, 128);
      }
    } else {
      const uint32* a32 =
          reinterpret_cast<const uint32*>(Ab + (size_t)(bm0 + ar) * K + kb + ac);
      uint32* ad = reinterpret_cast<uint32*>(&As[buf][ar][ac]);
#pragma unroll
      for (int j = 0; j < 4; ++j) ad[j] = a32[j];
      const uint32* b32 =
          reinterpret_cast<const uint32*>(Wt + (size_t)(bn0 + br) * K + kb + bc);
      uint32* bd = reinterpret_cast<uint32*>(&Bs[buf][br][bc]);
#pragma unroll
      for (int j = 0; j < 8; ++j) bd[j] = b32[j];
    }
  };

  // preload tile 0
  stage(0, 0);
#if __has_builtin(__builtin_amdgcn_s_wait_tensorcnt)
  if (kUseTdm && wave == 0) __builtin_amdgcn_s_wait_tensorcnt(0);
#endif
  __syncthreads();

  int buf = 0;
  for (int kb = 0; kb < K; kb += 32) {
    if (kb + 32 < K) stage(buf ^ 1, kb + 32);   // async DMA of next tile

    // A fragment: contiguous K order (low lanes K 0..15, high lanes K 16..31)
    v16bf afrag;
    {
      const __bf16* arow = &As[buf][waveM * 16 + (lane & 15)][0];
      const int kh = (lane < 16) ? 0 : 16;
#pragma unroll
      for (int j = 0; j < 8; ++j) afrag[j] = arow[kh + j];
#pragma unroll
      for (int j = 0; j < 8; ++j) afrag[8 + j] = arow[8 + kh + j];
    }

    // 4 back-to-back WMMAs sharing the A fragment; RA reuse hint is a
    // compile-time literal (required: builtin arg must be constant).
#define WMMA_NT(nt, RA)                                                        \
    {                                                                          \
      v16bf bfrag;                                                             \
      const __bf16* brow = &Bs[buf][waveN * 64 + (nt) * 16 + (lane & 15)][0];  \
      const int ko = (lane < 16) ? 0 : 16;                                     \
      _Pragma("unroll")                                                        \
      for (int j = 0; j < 16; ++j) bfrag[j] = brow[ko + j];                    \
      acc[nt] = __builtin_amdgcn_wmma_f32_16x16x32_bf16(                       \
          false, afrag, false, bfrag, (short)0, acc[nt], RA, false);           \
    }
    WMMA_NT(0, true)
    WMMA_NT(1, true)
    WMMA_NT(2, true)
    WMMA_NT(3, false)
#undef WMMA_NT

#if __has_builtin(__builtin_amdgcn_s_wait_tensorcnt)
    if (kUseTdm && wave == 0) __builtin_amdgcn_s_wait_tensorcnt(0);
#endif
    __syncthreads();   // next-tile DMA complete + this-tile reads complete
    buf ^= 1;
  }

  // epilogue: C/D layout -> VGPR r: lanes 0-15 row r, lanes 16-31 row r+8
  const int nbase = bn0 + waveN * 64 + (lane & 15);
  const int mbase = bm0 + waveM * 16 + ((lane < 16) ? 0 : 8);
#pragma unroll
  for (int nt = 0; nt < 4; ++nt) {
    const int n = nbase + nt * 16;
    const float bval = bias[n];
#pragma unroll
    for (int r = 0; r < 8; ++r) {
      const int m = mbase + r;
      float val = acc[nt][r] + bval;
      if (EPI == 1) val = 0.5f * val * (1.0f + erff(val * 0.70710678118654752f));
      if (EPI == 2) val += res[(size_t)m * N + n];
      if (OBF)
        reinterpret_cast<__bf16*>(Cout)[(size_t)m * N + n] = f2bf(val);
      else
        reinterpret_cast<float*>(Cout)[(size_t)m * N + n] = val;
    }
  }
}

// ---------------------------------------------------------------------------
// LayerNorm over last dim H (one row per block). OBF: bf16 or f32 output.
// ---------------------------------------------------------------------------
template <int OBF>
__global__ __launch_bounds__(256) void ln_kernel(
    const float* __restrict__ x, const float* __restrict__ g,
    const float* __restrict__ b, void* __restrict__ out, int H) {
  const int row = blockIdx.x;
  const int tid = threadIdx.x;
  const float* xr = x + (size_t)row * H;
  float s = 0.f, s2 = 0.f;
  for (int i = tid; i < H; i += 256) {
    float v = xr[i];
    s += v;
    s2 += v * v;
  }
  __shared__ float rs[256], rs2[256];
  rs[tid] = s; rs2[tid] = s2;
  __syncthreads();
  for (int off = 128; off > 0; off >>= 1) {
    if (tid < off) { rs[tid] += rs[tid + off]; rs2[tid] += rs2[tid + off]; }
    __syncthreads();
  }
  const float mu  = rs[0] / H;
  const float var = rs2[0] / H - mu * mu;
  const float inv = rsqrtf(var + 1e-12f);
  for (int i = tid; i < H; i += 256) {
    const float v = (xr[i] - mu) * inv * g[i] + b[i];
    if (OBF) reinterpret_cast<__bf16*>(out)[(size_t)row * H + i] = f2bf(v);
    else     reinterpret_cast<float*>(out)[(size_t)row * H + i] = v;
  }
}

// ---------------------------------------------------------------------------
// Attention mix: one wave32 per token. 17-way softmax over [self, E externals];
// gamma = doc_logprobs scales external values. Writes bf16 ctx for next GEMM.
// ---------------------------------------------------------------------------
__device__ __forceinline__ float wave_sum(float v) {
#pragma unroll
  for (int off = 16; off > 0; off >>= 1) v += __shfl_xor(v, off, 32);
  return v;
}

__global__ __launch_bounds__(256) void attn_kernel(
    const float* __restrict__ q, const float* __restrict__ kt,
    const float* __restrict__ vt, const float* __restrict__ kext,
    const float* __restrict__ vext, const float* __restrict__ dlp,
    __bf16* __restrict__ ctx) {
  const int wave = threadIdx.x >> 5;
  const int lane = threadIdx.x & 31;
  const int t = blockIdx.x * 8 + wave;   // token in [0, B*S)
  const int b = t >> 11;                 // S = 2048 tokens per batch

  const float* qp = q + (size_t)t * 768;
  float qreg[24];
#pragma unroll
  for (int i = 0; i < 24; ++i) qreg[i] = qp[lane + 32 * i];

  float sc[17];
  {
    const float* kp = kt + (size_t)t * 768;
    float p = 0.f;
#pragma unroll
    for (int i = 0; i < 24; ++i) p += qreg[i] * kp[lane + 32 * i];
    sc[0] = wave_sum(p);
  }
  for (int e = 0; e < 16; ++e) {
    const float* kp = kext + ((size_t)(b * 16 + e)) * 768;
    float p = 0.f;
#pragma unroll
    for (int i = 0; i < 24; ++i) p += qreg[i] * kp[lane + 32 * i];
    sc[1 + e] = wave_sum(p);
  }

  float mx = sc[0];
#pragma unroll
  for (int j = 1; j < 17; ++j) mx = fmaxf(mx, sc[j]);
  float sum = 0.f;
#pragma unroll
  for (int j = 0; j < 17; ++j) { sc[j] = expf(sc[j] - mx); sum += sc[j]; }
  const float inv = 1.f / sum;
  const float w0 = sc[0] * inv;
  float we[16];
#pragma unroll
  for (int e = 0; e < 16; ++e) we[e] = sc[1 + e] * inv * dlp[b * 16 + e];

  const float* vp = vt + (size_t)t * 768;
#pragma unroll
  for (int i = 0; i < 24; ++i) {
    const int h = lane + 32 * i;
    float a = w0 * vp[h];
#pragma unroll
    for (int e = 0; e < 16; ++e)
      a += we[e] * vext[((size_t)(b * 16 + e)) * 768 + h];
    ctx[(size_t)t * 768 + h] = f2bf(a);
  }
}

// ---------------------------------------------------------------------------
extern "C" void kernel_launch(void* const* d_in, const int* in_sizes, int n_in,
                              void* d_out, int out_size, void* d_ws, size_t ws_size,
                              hipStream_t stream) {
  (void)in_sizes; (void)n_in; (void)out_size; (void)ws_size;
  const float* hidden = (const float*)d_in[0];
  const float* extemb = (const float*)d_in[1];
  const float* dlp    = (const float*)d_in[2];
  const float* Wq = (const float*)d_in[3];  const float* bq = (const float*)d_in[4];
  const float* Wk = (const float*)d_in[5];  const float* bk = (const float*)d_in[6];
  const float* Wv = (const float*)d_in[7];  const float* bv = (const float*)d_in[8];
  const float* Wo = (const float*)d_in[9];  const float* bo = (const float*)d_in[10];
  const float* ln_g = (const float*)d_in[11]; const float* ln_b = (const float*)d_in[12];
  const float* W1 = (const float*)d_in[13]; const float* b1 = (const float*)d_in[14];
  const float* W2 = (const float*)d_in[15]; const float* b2 = (const float*)d_in[16];
  const float* mlp_g = (const float*)d_in[17]; const float* mlp_b = (const float*)d_in[18];

  constexpr int B = 8, S = 2048, H = 768, E = 16, I = 3072;
  constexpr int T = B * S;   // 16384
  constexpr int R = B * E;   // 128

  // ---- workspace carve ----
  char* p = (char*)d_ws;
  auto carveF = [&p](size_t n) { float* r = (float*)p; p += n * sizeof(float); return r; };
  auto carveB = [&p](size_t n) { __bf16* r = (__bf16*)p; p += n * sizeof(__bf16); return r; };
  float* q    = carveF((size_t)T * H);
  float* k    = carveF((size_t)T * H);   // reused as pre-LN tmp
  float* v    = carveF((size_t)T * H);
  float* extp = carveF((size_t)R * H);
  float* kext = carveF((size_t)R * H);
  float* vext = carveF((size_t)R * H);
  __bf16* hid_b = carveB((size_t)T * H);
  __bf16* Wq_t  = carveB((size_t)H * H);   // transposed [N][K] bf16
  __bf16* Wk_t  = carveB((size_t)H * H);
  __bf16* Wv_t  = carveB((size_t)H * H);
  __bf16* Wo_t  = carveB((size_t)H * H);
  __bf16* W1_t  = carveB((size_t)H * I);   // [I][H]
  __bf16* W2_t  = carveB((size_t)I * H);   // [H][I]
  __bf16* xe_b  = carveB((size_t)R * H);
  __bf16* h1_b  = carveB((size_t)R * I);
  __bf16* ext_b = carveB((size_t)R * H);
  __bf16* ctx_b = carveB((size_t)T * H);
  float* tmp = k;

  const dim3 blk(256);
  auto cvt = [&](const float* src, __bf16* dst, size_t n) {
    const int n4 = (int)(n / 4);
    to_bf16_kernel<<<dim3((n4 + 255) / 256), blk, 0, stream>>>(src, dst, n4);
  };
  auto tr = [&](const float* W, __bf16* Wt, int K, int N) {
    transpose_to_bf16<<<dim3(N / 32, K / 32), blk, 0, stream>>>(W, Wt, K, N);
  };

  // ---- one-time quantize (A-side) and transpose+quantize (B-side) ----
  cvt(hidden, hid_b, (size_t)T * H);
  cvt(extemb, xe_b, (size_t)R * H);
  tr(Wq, Wq_t, H, H);  tr(Wk, Wk_t, H, H);
  tr(Wv, Wv_t, H, H);  tr(Wo, Wo_t, H, H);
  tr(W1, W1_t, H, I);  tr(W2, W2_t, I, H);

  // ---- external-embedding MLP block: gelu(x@W1+b1)@W2+b2 (+res) -> LN ----
  gemm_bf16_wmma<1, 1><<<dim3(I / 128, R / 64), blk, 0, stream>>>(xe_b, W1_t, b1, nullptr, h1_b, R, I, H);
  gemm_bf16_wmma<2, 0><<<dim3(H / 128, R / 64), blk, 0, stream>>>(h1_b, W2_t, b2, extemb, extp, R, H, I);
  ln_kernel<1><<<dim3(R), blk, 0, stream>>>(extp, mlp_g, mlp_b, ext_b, H);

  // ---- projections ----
  gemm_bf16_wmma<0, 0><<<dim3(H / 128, T / 64), blk, 0, stream>>>(hid_b, Wq_t, bq, nullptr, q, T, H, H);
  gemm_bf16_wmma<0, 0><<<dim3(H / 128, T / 64), blk, 0, stream>>>(hid_b, Wk_t, bk, nullptr, k, T, H, H);
  gemm_bf16_wmma<0, 0><<<dim3(H / 128, T / 64), blk, 0, stream>>>(hid_b, Wv_t, bv, nullptr, v, T, H, H);
  gemm_bf16_wmma<0, 0><<<dim3(H / 128, R / 64), blk, 0, stream>>>(ext_b, Wk_t, bk, nullptr, kext, R, H, H);
  gemm_bf16_wmma<0, 0><<<dim3(H / 128, R / 64), blk, 0, stream>>>(ext_b, Wv_t, bv, nullptr, vext, R, H, H);

  // ---- per-token 17-way attention mix -> bf16 ctx ----
  attn_kernel<<<dim3(T / 8), blk, 0, stream>>>(q, k, v, kext, vext, dlp, ctx_b);

  // ---- output dense + residual, then LayerNorm into d_out ----
  gemm_bf16_wmma<2, 0><<<dim3(H / 128, T / 64), blk, 0, stream>>>(ctx_b, Wo_t, bo, hidden, tmp, T, H, H);
  ln_kernel<0><<<dim3(T), blk, 0, stream>>>(tmp, ln_g, ln_b, d_out, H);
}
